// PointNet_89232240542472
// MI455X (gfx1250) — matrix operand order repaired and checked by reading, not compiled, not run on hardware
//
#include <hip/hip_runtime.h>

typedef __attribute__((ext_vector_type(16))) _Float16 v16h;
typedef __attribute__((ext_vector_type(8)))  float    v8f;

// ---------------- problem constants ----------------
#define BB    4
#define NN    8192
#define SNP   1024          // NPTS
#define KNB   48            // K neighbors per centroid
#define KCC   6             // KC
#define BN    (BB*NN)       // 32768 point rows
#define CHC   512           // centroids per chunk
#define NCHUNK ((BB*SNP)/CHC)
#define RCH   (CHC*KNB)     // 24576 group rows per chunk

__device__ __forceinline__ float sigmoidf_(float v){ return 1.f/(1.f+__expf(-v)); }

// =====================================================================
// WMMA GEMM:  C[M,N] = epilogue(A[M,K] @ W[K,N] + bias)
// Wt = f16 weight, transposed: Wt[n*ldWt + k].
// Block = 128 threads (4 waves). Block tile = 64 rows x (TNT*16) cols.
// Weight tile (32 x TNT*16 f16) is block-uniform, so it is staged through
// LDS once per block with double buffering (removes the 4x redundant
// global reads the naive version did). Each wave owns a 16-row strip and
// TNT 16x16 f32 accumulators; K stepped by 32 (v_wmma_f32_16x16x32_f16).
// A fragment: lane l(=lane&15) holds row M=l; hi=lane>>4
// selects K run hi*16..+15 (four contiguous float4 loads), same order as B.
// B fragment: lane l holds column n, halves j -> K = hi*16 + j, i.e. one
// contiguous 32B run in the transposed weight -> one LDS "run" slot.
// ACT: 0 none, 1 relu, 2 leaky(0.01). ACCUM: C += . GATE: C += sigmoid(v)*G2
// =====================================================================
template<int ACT, int ACCUM, int GATE, int TNT>
__global__ void __launch_bounds__(128)
gemm_wmma_k(const float* __restrict__ A, const _Float16* __restrict__ Wt,
            const float* __restrict__ bias, float* __restrict__ C,
            int M, int N, int K, int ldWt, const float* __restrict__ G2)
{
  (void)M;
  __shared__ _Float16 Bs[2][TNT*512];     // TNT*32 runs of 16 halves per buffer

  const int tid  = threadIdx.x;
  const int lane = tid & 31;
  const int wv   = tid >> 5;
  const int l    = lane & 15;
  const int hi   = lane >> 4;
  const int colBase = blockIdx.y * (TNT*16);

  const long rowA = (long)blockIdx.x*64 + wv*16 + l;
  const float* Ap = A + rowA*(long)K;
  const int nsteps = K >> 5;

  // cooperative weight-tile loader: run r -> col = r>>1, K-half = r&1
  auto loadB = [&](int buf, int k0) {
#pragma unroll
    for (int r0 = 0; r0 < TNT*32; r0 += 128) {
      const int r   = r0 + tid;
      const int col = r >> 1, hk = r & 1;
      const _Float16* src = Wt + (long)(colBase + col)*ldWt + k0 + hk*16;
      *reinterpret_cast<v16h*>(&Bs[buf][r*16]) = *reinterpret_cast<const v16h*>(src);
    }
  };

  loadB(0, 0);
  __syncthreads();

  v8f acc[TNT] = {};
  for (int s = 0; s < nsteps; ++s) {
    const int k0 = s << 5;
    if (s + 1 < nsteps) loadB((s+1)&1, k0+32);     // prefetch next tile into other buffer

    // A fragment: one contiguous 16-float run (K = hi*16 + j), converted to f16 on the fly
    const float* ap = Ap + k0 + hi*16;
    if (s + 1 < nsteps) __builtin_prefetch((const void*)(ap + 32), 0, 0);
    float4 a0 = *reinterpret_cast<const float4*>(ap);
    float4 a1 = *reinterpret_cast<const float4*>(ap + 4);
    float4 a2 = *reinterpret_cast<const float4*>(ap + 8);
    float4 a3 = *reinterpret_cast<const float4*>(ap + 12);
    v16h af;
    af[0]=(_Float16)a0.x;  af[1]=(_Float16)a0.y;  af[2]=(_Float16)a0.z;  af[3]=(_Float16)a0.w;
    af[4]=(_Float16)a1.x;  af[5]=(_Float16)a1.y;  af[6]=(_Float16)a1.z;  af[7]=(_Float16)a1.w;
    af[8]=(_Float16)a2.x;  af[9]=(_Float16)a2.y;  af[10]=(_Float16)a2.z; af[11]=(_Float16)a2.w;
    af[12]=(_Float16)a3.x; af[13]=(_Float16)a3.y; af[14]=(_Float16)a3.z; af[15]=(_Float16)a3.w;

    const _Float16* bb = &Bs[s & 1][0];
#pragma unroll
    for (int t = 0; t < TNT; ++t) {
      const int run = (t*16 + l)*2 + hi;
      v16h bf = *reinterpret_cast<const v16h*>(bb + run*16);
      acc[t] = __builtin_amdgcn_wmma_f32_16x16x32_f16(false, af, false, bf,
                                                      (short)0, acc[t], false, false);
    }
    __syncthreads();   // reads of buf s&1 done before it is overwritten at s+2
  }

  const long rowC = (long)blockIdx.x*64 + wv*16 + hi*8;
#pragma unroll
  for (int t = 0; t < TNT; ++t) {
    const int cc = colBase + t*16 + l;
    const float bz = bias ? bias[cc] : 0.f;
#pragma unroll
    for (int r = 0; r < 8; ++r) {
      const long idx = (rowC + r)*(long)N + cc;
      float v = acc[t][r] + bz;
      if (ACCUM) v += C[idx];
      if (ACT == 1) v = fmaxf(v, 0.f);
      if (ACT == 2) v = (v > 0.f) ? v : 0.01f*v;
      if (GATE)  C[idx] = C[idx] + sigmoidf_(v) * G2[idx];
      else       C[idx] = v;
    }
  }
}

// ---------------- f32 -> f16 transposed weight conversion ----------------
__global__ void cvt_t_k(const float* __restrict__ W, _Float16* __restrict__ Wt,
                        int Kd, int Nd)
{
  int t = blockIdx.x*blockDim.x + threadIdx.x;
  if (t >= Kd*Nd) return;
  int k = t % Kd, n = t / Kd;
  Wt[(long)n*Kd + k] = (_Float16)W[(long)k*Nd + n];
}

// ---------------- LayerNorm (+optional relu), in place, 1 block per row ----
__global__ void layernorm_act_k(float* __restrict__ X, const float* __restrict__ g,
                                const float* __restrict__ b, int D, int doRelu)
{
  const long row = blockIdx.x;
  float* x = X + row*(long)D;
  __shared__ float s1[256], s2[256];
  float a = 0.f, q = 0.f;
  for (int i = threadIdx.x; i < D; i += blockDim.x) { float v = x[i]; a += v; q += v*v; }
  s1[threadIdx.x] = a; s2[threadIdx.x] = q; __syncthreads();
  for (int st = blockDim.x >> 1; st > 0; st >>= 1) {
    if ((int)threadIdx.x < st) { s1[threadIdx.x] += s1[threadIdx.x+st]; s2[threadIdx.x] += s2[threadIdx.x+st]; }
    __syncthreads();
  }
  const float mean = s1[0] / (float)D;
  const float var  = s2[0] / (float)D - mean*mean;
  const float inv  = rsqrtf(var + 1e-5f);
  for (int i = threadIdx.x; i < D; i += blockDim.x) {
    float v = (x[i]-mean)*inv*g[i] + b[i];
    if (doRelu) v = fmaxf(v, 0.f);
    x[i] = v;
  }
}

// ---------------- point embed: x = [pvdata(:4), gelu(xyz@w1+b1)@w2+b2] ------
__global__ void point_embed_k(const float* __restrict__ pv,
                              const float* __restrict__ w1, const float* __restrict__ b1,
                              const float* __restrict__ w2, const float* __restrict__ b2,
                              float* __restrict__ x)
{
  long p = (long)blockIdx.x*blockDim.x + threadIdx.x;
  if (p >= BN) return;
  const float* q = pv + p*4;
  float i0 = q[0], i1 = q[1], i2 = q[2];
  float h[32];
#pragma unroll
  for (int j = 0; j < 32; ++j) {
    float v = i0*w1[0*32+j] + i1*w1[1*32+j] + i2*w1[2*32+j] + b1[j];
    h[j] = 0.5f*v*(1.f + erff(v*0.70710678118654752f));   // exact gelu
  }
  float* xr = x + p*64;
  xr[0]=q[0]; xr[1]=q[1]; xr[2]=q[2]; xr[3]=q[3];
  for (int j = 0; j < 60; ++j) {
    float v = b2[j];
#pragma unroll
    for (int k = 0; k < 32; ++k) v += h[k]*w2[k*60+j];
    xr[4+j] = v;
  }
}

// ---------------- per-batch column max over rowsPerB rows -------------------
__global__ void colmax_k(const float* __restrict__ X, float* __restrict__ out,
                         int rowsPerB, int D)
{
  int b = blockIdx.x;
  int f = blockIdx.y*blockDim.x + threadIdx.x;
  if (f >= D) return;
  const float* Xb = X + (long)b*rowsPerB*D + f;
  float m = -3.4e38f;
  for (long r = 0; r < rowsPerB; ++r) m = fmaxf(m, Xb[r*(long)D]);
  out[(long)b*D + f] = m;
}

// ---------------- out = [broadcast(max_b), X] over feature dim ------------
__global__ void concat_bcast_k(const float* __restrict__ mx, const float* __restrict__ X,
                               float* __restrict__ out, int rowsPerB, int D, long total)
{
  long t = (long)blockIdx.x*blockDim.x + threadIdx.x;
  if (t >= total) return;
  int D2 = 2*D;
  long r = t / D2; int d = (int)(t - r*D2);
  int b = (int)(r / rowsPerB);
  out[t] = (d < D) ? mx[(long)b*D + d] : X[r*(long)D + (d - D)];
}

// ---------------- farthest point sampling (sequential scan) ----------------
__global__ void __launch_bounds__(1024)
fps_k(const float* __restrict__ pv, int* __restrict__ cidx, float* __restrict__ cxyz)
{
  const int b = blockIdx.x, tid = threadIdx.x;
  const float* P = pv + (long)b*NN*4;
  float px[8], py[8], pz[8], dd[8];
#pragma unroll
  for (int i = 0; i < 8; ++i) {
    int p = i*1024 + tid;
    px[i] = P[p*4+0]; py[i] = P[p*4+1]; pz[i] = P[p*4+2]; dd[i] = 1e10f;
  }
  __shared__ float sf[3];
  __shared__ float sm[1024];
  __shared__ int   si[1024];
  int far = 0;
  for (int s = 0; s < SNP; ++s) {
    if (tid == 0) {
      cidx[b*SNP + s] = far;
      float fx = P[far*4], fy = P[far*4+1], fz = P[far*4+2];
      sf[0] = fx; sf[1] = fy; sf[2] = fz;
      cxyz[((long)b*SNP + s)*3 + 0] = fx;
      cxyz[((long)b*SNP + s)*3 + 1] = fy;
      cxyz[((long)b*SNP + s)*3 + 2] = fz;
    }
    __syncthreads();
    float fx = sf[0], fy = sf[1], fz = sf[2];
    float best = -1.f; int bi = 0;
#pragma unroll
    for (int i = 0; i < 8; ++i) {
      float dx = px[i]-fx, dy = py[i]-fy, dz = pz[i]-fz;
      float d = dx*dx + dy*dy + dz*dz;
      dd[i] = fminf(dd[i], d);
      if (dd[i] > best) { best = dd[i]; bi = i*1024 + tid; }
    }
    sm[tid] = best; si[tid] = bi; __syncthreads();
    for (int st = 512; st > 0; st >>= 1) {
      if (tid < st) {
        if (sm[tid+st] > sm[tid] || (sm[tid+st] == sm[tid] && si[tid+st] < si[tid])) {
          sm[tid] = sm[tid+st]; si[tid] = si[tid+st];
        }
      }
      __syncthreads();
    }
    far = si[0];
  }
}

// ---------------- top-48 nearest points per centroid (LDS argmin sweep) ----
__global__ void knn_k(const float* __restrict__ pv, const float* __restrict__ cxyz,
                      int* __restrict__ kidx, float* __restrict__ kdist)
{
  const int c = blockIdx.x;            // 0 .. BB*SNP-1
  const int b = c / SNP;
  const float* P = pv + (long)b*NN*4;
  const float cx = cxyz[c*3+0], cy = cxyz[c*3+1], cz = cxyz[c*3+2];
  extern __shared__ float sdist[];     // NN floats (32 KB)
  for (int i = threadIdx.x; i < NN; i += 256) {
    float dx = P[i*4]-cx, dy = P[i*4+1]-cy, dz = P[i*4+2]-cz;
    sdist[i] = dx*dx + dy*dy + dz*dz;
  }
  __syncthreads();
  __shared__ float smn[256];
  __shared__ int   smi[256];
  for (int k = 0; k < KNB; ++k) {
    float best = 3.4e38f; int bi = 0;
    for (int i = threadIdx.x; i < NN; i += 256) {
      float d = sdist[i];
      if (d < best) { best = d; bi = i; }
    }
    smn[threadIdx.x] = best; smi[threadIdx.x] = bi; __syncthreads();
    for (int st = 128; st > 0; st >>= 1) {
      if ((int)threadIdx.x < st) {
        if (smn[threadIdx.x+st] < smn[threadIdx.x] ||
            (smn[threadIdx.x+st] == smn[threadIdx.x] && smi[threadIdx.x+st] < smi[threadIdx.x])) {
          smn[threadIdx.x] = smn[threadIdx.x+st]; smi[threadIdx.x] = smi[threadIdx.x+st];
        }
      }
      __syncthreads();
    }
    if (threadIdx.x == 0) {
      int w = smi[0];
      kidx[(long)c*KNB + k]  = w;
      float sq = smn[0];
      kdist[(long)c*KNB + k] = (sq > 0.f) ? sqrtf(sq) : 0.f;   // safe_sqrt
      sdist[w] = 3.4e38f;
    }
    __syncthreads();
  }
}

// ---------------- gather grouped features gp = [pvdata4, dc] ---------------
__global__ void build_gp_k(const float* __restrict__ pv, const int* __restrict__ kidx,
                           const float* __restrict__ kdist, float* __restrict__ gp, int c0)
{
  int t = blockIdx.x*blockDim.x + threadIdx.x;
  if (t >= RCH) return;
  int ci = c0 + t / KNB;
  int k  = t - (t / KNB)*KNB;
  int b  = ci / SNP;
  int p  = kidx[(long)ci*KNB + k];
  const float* q = pv + ((long)b*NN + p)*4;
  float* o = gp + (long)t*5;
  o[0]=q[0]; o[1]=q[1]; o[2]=q[2]; o[3]=q[3];
  o[4]=kdist[(long)ci*KNB + k];
}

// ---------------- tiny MLP 5 -> 64 (gp1 first layer) ----------------------
__global__ void mlp5_64_k(const float* __restrict__ gp, const float* __restrict__ w,
                          const float* __restrict__ b, float* __restrict__ out, int rows)
{
  long t = (long)blockIdx.x*blockDim.x + threadIdx.x;
  if (t >= (long)rows*64) return;
  long r = t >> 6; int j = (int)(t & 63);
  const float* in = gp + r*5;
  float v = b[j];
#pragma unroll
  for (int i = 0; i < 5; ++i) v += in[i]*w[i*64 + j];
  out[r*64 + j] = v;
}

// ---------------- max over K within group + concat(bcast, X) ---------------
__global__ void gmax_concat_k(const float* __restrict__ h, float* __restrict__ hc,
                              int nc, int Kn, int D)
{
  int t = blockIdx.x*blockDim.x + threadIdx.x;
  if (t >= nc*D) return;
  int c = t / D, d = t - (t/D)*D;
  const float* base = h + ((long)c*Kn)*D + d;
  float m = -3.4e38f;
  for (int k = 0; k < Kn; ++k) m = fmaxf(m, base[(long)k*D]);
  float* ob = hc + ((long)c*Kn)*2*D + d;
  for (int k = 0; k < Kn; ++k) {
    ob[(long)k*2*D]     = m;
    ob[(long)k*2*D + D] = base[(long)k*D];
  }
}

__global__ void gmax_k(const float* __restrict__ h, float* __restrict__ out,
                       int nc, int Kn, int D)
{
  int t = blockIdx.x*blockDim.x + threadIdx.x;
  if (t >= nc*D) return;
  int c = t / D, d = t - (t/D)*D;
  const float* base = h + ((long)c*Kn)*D + d;
  float m = -3.4e38f;
  for (int k = 0; k < Kn; ++k) m = fmaxf(m, base[(long)k*D]);
  out[(long)c*D + d] = m;
}

// ---------------- top-6 centroids per point + softmax attn -----------------
__global__ void knc_attn_k(const float* __restrict__ pv, const float* __restrict__ cxyz,
                           float* __restrict__ attn, int* __restrict__ idx2)
{
  long t = (long)blockIdx.x*256 + threadIdx.x;      // point id, < BN
  int b = (int)(t / NN);
  __shared__ float sc[SNP*3];
  for (int i = threadIdx.x; i < SNP*3; i += 256) sc[i] = cxyz[(long)b*SNP*3 + i];
  __syncthreads();
  float x = pv[t*4], y = pv[t*4+1], z = pv[t*4+2];
  float bd[KCC]; int bi[KCC];
#pragma unroll
  for (int j = 0; j < KCC; ++j) { bd[j] = 3.4e38f; bi[j] = 0; }
  for (int c = 0; c < SNP; ++c) {
    float dx = x-sc[c*3], dy = y-sc[c*3+1], dz = z-sc[c*3+2];
    float d = dx*dx + dy*dy + dz*dz;
    if (d < bd[KCC-1]) {
      int p = KCC-1;
      while (p > 0 && d < bd[p-1]) { bd[p] = bd[p-1]; bi[p] = bi[p-1]; --p; }
      bd[p] = d; bi[p] = c;
    }
  }
  float nv[KCC], mx = -3.4e38f;
#pragma unroll
  for (int j = 0; j < KCC; ++j) {
    float s = (bd[j] > 0.f) ? sqrtf(bd[j]) : 0.f;  // safe_sqrt
    nv[j] = -s; mx = fmaxf(mx, nv[j]);
  }
  float ssum = 0.f;
#pragma unroll
  for (int j = 0; j < KCC; ++j) { nv[j] = __expf(nv[j]-mx); ssum += nv[j]; }
  float inv = 1.f/ssum;
#pragma unroll
  for (int j = 0; j < KCC; ++j) { attn[t*KCC + j] = nv[j]*inv; idx2[t*KCC + j] = bi[j]; }
}

// ---------------- local = sum_j attn_j * pvgW[idx2_j] ----------------------
__global__ void local_gather_k(const float* __restrict__ pvgW, const float* __restrict__ attn,
                               const int* __restrict__ idx2, float* __restrict__ local)
{
  long t = (long)blockIdx.x*blockDim.x + threadIdx.x;
  if (t >= (long)BN*512) return;
  int d = (int)(t & 511);
  long p = t >> 9;
  int b = (int)(p / NN);
  const float* aw = attn + p*KCC;
  const int*   iw = idx2 + p*KCC;
  float v = 0.f;
#pragma unroll
  for (int j = 0; j < KCC; ++j)
    v += aw[j] * pvgW[((long)b*SNP + iw[j])*512 + d];
  local[t] = v;
}

// ---------------- gctx = [broadcast(g), f2] --------------------------------
__global__ void gctx_k(const float* __restrict__ g, const float* __restrict__ f2,
                       float* __restrict__ gctx)
{
  long t = (long)blockIdx.x*blockDim.x + threadIdx.x;
  if (t >= (long)BN*1024) return;
  int d = (int)(t & 1023);
  long p = t >> 10;
  int b = (int)(p / NN);
  gctx[t] = (d < 512) ? g[(long)b*512 + d] : f2[p*512 + (d-512)];
}

// ---------------- head: out = o(64) @ lg_w + lg_b --------------------------
__global__ void head_k(const float* __restrict__ o, const float* __restrict__ w,
                       const float* __restrict__ b, float* __restrict__ out)
{
  long t = (long)blockIdx.x*blockDim.x + threadIdx.x;
  if (t >= (long)BN*3) return;
  long r = t / 3; int j = (int)(t - r*3);
  const float* x = o + r*64;
  float v = b[j];
#pragma unroll
  for (int i = 0; i < 64; ++i) v += x[i]*w[i*3 + j];
  out[t] = v;
}

// =====================================================================
// host side
// =====================================================================
static inline long cdiv_(long a, long b) { return (a + b - 1) / b; }

// sorted param order (pytree dict flattening sorts keys)
enum {
  P_ATTN_W = 0, P_FG_B1, P_FG_B2, P_FG_W1, P_FG_W2,
  P_FLP_B, P_FLP_BE, P_FLP_G, P_FLP_W,
  P_GE1_B1, P_GE1_B2, P_GE1_BE, P_GE1_G, P_GE1_W1, P_GE1_W2,
  P_GE2_B1, P_GE2_B2, P_GE2_BE, P_GE2_G, P_GE2_W1, P_GE2_W2,
  P_GP1_B1, P_GP1_B2, P_GP1_BE, P_GP1_G, P_GP1_W1, P_GP1_W2,
  P_GP2_B1, P_GP2_B2, P_GP2_BE, P_GP2_G, P_GP2_W1, P_GP2_W2,
  P_L4_B, P_L4_BE, P_L4_G, P_L4_W,
  P_L5_B, P_L5_W, P_L6_B, P_L6_W, P_L7_B, P_L7_W,
  P_LG_B, P_LG_W, P_PE_B1, P_PE_B2, P_PE_W1, P_PE_W2
};

// f16 transposed weight arena offsets (halves; all 32B-aligned)
#define WT_GE1_1 0u
#define WT_GE1_2 4096u
#define WT_GE2_1 12288u
#define WT_GE2_2 143360u
#define WT_GP1_2 405504u
#define WT_GP2_1 413696u
#define WT_GP2_2 544768u
#define WT_ATTN  806912u
#define WT_FLP   1069056u
#define WT_FG1   1593344u
#define WT_FG2   1724416u
#define WT_L4    1789952u
#define WT_L5    2838528u
#define WT_L6    3362816u
#define WT_L7    3493888u
#define WT_TOTAL 3510272u

extern "C" void kernel_launch(void* const* d_in, const int* in_sizes, int n_in,
                              void* d_out, int out_size, void* d_ws, size_t ws_size,
                              hipStream_t stream)
{
  (void)in_sizes; (void)n_in; (void)out_size; (void)ws_size;
  const float* pv = (const float*)d_in[1];                 // pvdata (B,N,4); mdata unused
  const float* pr[49];
  for (int i = 0; i < 49; ++i) pr[i] = (const float*)d_in[2 + i];
  float* outp = (float*)d_out;

  // ---- workspace bump allocator (float units, 256B aligned) ----
  float* base = (float*)d_ws;
  size_t fo = 0;
  auto A = [&](size_t n) -> float* {
    float* p = base + fo; fo += (n + 63) & ~(size_t)63; return p;
  };
  const size_t SCRATCH_F = 36306944;                       // chunk-stage peak
  float* SC    = A(SCRATCH_F);
  float* f2    = A((size_t)BN*512);
  float* gctx  = A((size_t)BN*1024);
  float* la    = A((size_t)BN*1024);
  float* pvg   = A((size_t)BB*SNP*512);
  float* pvgW  = A((size_t)BB*SNP*512);
  float* attn  = A((size_t)BN*KCC);
  float* kdist = A((size_t)BB*SNP*KNB);
  float* cxyz  = A((size_t)BB*SNP*3);
  float* gmx   = A(BB*512);
  float* f1g   = A(BB*128);
  int*   idx2  = (int*)A((size_t)BN*KCC);
  int*   kidx  = (int*)A((size_t)BB*SNP*KNB);
  int*   cidx  = (int*)A(BB*SNP);
  _Float16* wt = (_Float16*)A(WT_TOTAL/2 + 64);

  // scratch overlays
  float* x_  = SC;                                   // BN*64
  float* t1  = SC + 2097152;                         // BN*64
  float* f1  = SC + 4194304;                         // BN*128
  float* f1c = SC + 8388608;                         // BN*256
  float* t2  = SC + 16777216;                        // BN*512
  float* gpb = SC;                                   // RCH*5
  float* u1  = SC + 131072;                          // RCH*64
  float* h1  = SC + 1703936;                         // RCH*128
  float* h1c = SC + 4849664;                         // RCH*256
  float* v2  = SC + 11141120;                        // RCH*512
  float* h2  = SC + 23724032;                        // RCH*512
  float* lcl = SC;                                   // BN*512
  float* tg  = SC + 16777216;                        // BN*64

  // tile-N dispatch: 128-wide strips when possible (halves A re-reads),
  // 64-wide for N==64 layers
  #define GEMM(ACT, ACC, GT, A_, W_, B_, C_, M_, N_, K_, LD_, G2_)                         \
    do {                                                                                   \
      if ((N_) % 128 == 0)                                                                 \
        gemm_wmma_k<ACT,ACC,GT,8><<<dim3((unsigned)((M_)/64),(unsigned)((N_)/128)),        \
                                    128, 0, stream>>>(A_, W_, B_, C_, M_, N_, K_, LD_, G2_);\
      else                                                                                 \
        gemm_wmma_k<ACT,ACC,GT,4><<<dim3((unsigned)((M_)/64),(unsigned)((N_)/64)),         \
                                    128, 0, stream>>>(A_, W_, B_, C_, M_, N_, K_, LD_, G2_);\
    } while (0)

  // ---- convert + transpose all WMMA weights to f16 ----
  struct Job { const float* W; unsigned off; int K; int N; };
  const Job jobs[] = {
    { pr[P_GE1_W1], WT_GE1_1,   64,   64 },
    { pr[P_GE1_W2], WT_GE1_2,   64,  128 },
    { pr[P_GE2_W1], WT_GE2_1,  256,  512 },
    { pr[P_GE2_W2], WT_GE2_2,  512,  512 },
    { pr[P_GP1_W2], WT_GP1_2,   64,  128 },
    { pr[P_GP2_W1], WT_GP2_1,  256,  512 },
    { pr[P_GP2_W2], WT_GP2_2,  512,  512 },
    { pr[P_ATTN_W], WT_ATTN,   512,  512 },
    { pr[P_FLP_W],  WT_FLP,    512, 1024 },
    { pr[P_FG_W1],  WT_FG1,   2048,   64 },
    { pr[P_FG_W2],  WT_FG2,     64, 1024 },
    { pr[P_L4_W],   WT_L4,    1024, 1024 },
    { pr[P_L5_W],   WT_L5,    1024,  512 },
    { pr[P_L6_W],   WT_L6,     512,  256 },
    { pr[P_L7_W],   WT_L7,     256,   64 },
  };
  for (const Job& j : jobs)
    cvt_t_k<<<(unsigned)cdiv_((long)j.K*j.N, 256), 256, 0, stream>>>(j.W, wt + j.off, j.K, j.N);

  // ---- stage A: per-point encoder ----
  point_embed_k<<<(unsigned)cdiv_(BN,256), 256, 0, stream>>>(
      pv, pr[P_PE_W1], pr[P_PE_B1], pr[P_PE_W2], pr[P_PE_B2], x_);
  GEMM(0,0,0, x_,  wt+WT_GE1_1, pr[P_GE1_B1], t1, BN,  64,  64,  64, nullptr);
  layernorm_act_k<<<BN, 256, 0, stream>>>(t1, pr[P_GE1_G], pr[P_GE1_BE], 64, 1);
  GEMM(0,0,0, t1,  wt+WT_GE1_2, pr[P_GE1_B2], f1, BN, 128,  64,  64, nullptr);
  colmax_k<<<dim3(BB,1), 256, 0, stream>>>(f1, f1g, NN, 128);
  concat_bcast_k<<<(unsigned)cdiv_((long)BN*256,256), 256, 0, stream>>>(
      f1g, f1, f1c, NN, 128, (long)BN*256);
  GEMM(0,0,0, f1c, wt+WT_GE2_1, pr[P_GE2_B1], t2, BN, 512, 256, 256, nullptr);
  layernorm_act_k<<<BN, 256, 0, stream>>>(t2, pr[P_GE2_G], pr[P_GE2_BE], 512, 1);
  GEMM(0,0,0, t2,  wt+WT_GE2_2, pr[P_GE2_B2], f2, BN, 512, 512, 512, nullptr);
  colmax_k<<<dim3(BB,2), 256, 0, stream>>>(f2, gmx, NN, 512);

  // ---- stage B: FPS + kNN ----
  fps_k<<<BB, 1024, 0, stream>>>(pv, cidx, cxyz);
  knn_k<<<BB*SNP, 256, NN*sizeof(float), stream>>>(pv, cxyz, kidx, kdist);

  // ---- stage C: grouped PointNet, chunked over centroids ----
  for (int ch = 0; ch < NCHUNK; ++ch) {
    int c0 = ch*CHC;
    build_gp_k<<<(unsigned)cdiv_(RCH,256), 256, 0, stream>>>(pv, kidx, kdist, gpb, c0);
    mlp5_64_k<<<(unsigned)cdiv_((long)RCH*64,256), 256, 0, stream>>>(
        gpb, pr[P_GP1_W1], pr[P_GP1_B1], u1, RCH);
    layernorm_act_k<<<RCH, 256, 0, stream>>>(u1, pr[P_GP1_G], pr[P_GP1_BE], 64, 1);
    GEMM(0,0,0, u1,  wt+WT_GP1_2, pr[P_GP1_B2], h1, RCH, 128,  64,  64, nullptr);
    gmax_concat_k<<<(unsigned)cdiv_((long)CHC*128,256), 256, 0, stream>>>(h1, h1c, CHC, KNB, 128);
    GEMM(0,0,0, h1c, wt+WT_GP2_1, pr[P_GP2_B1], v2, RCH, 512, 256, 256, nullptr);
    layernorm_act_k<<<RCH, 256, 0, stream>>>(v2, pr[P_GP2_G], pr[P_GP2_BE], 512, 1);
    GEMM(0,0,0, v2,  wt+WT_GP2_2, pr[P_GP2_B2], h2, RCH, 512, 512, 512, nullptr);
    gmax_k<<<(unsigned)cdiv_((long)CHC*512,256), 256, 0, stream>>>(
        h2, pvg + (size_t)c0*512, CHC, KNB, 512);
  }

  // ---- stage D: attention + fusion + head ----
  // pvgW = pvg @ attn_w  (gather AFTER matmul: exact reorder of cf)
  GEMM(0,0,0, pvg, wt+WT_ATTN, nullptr, pvgW, BB*SNP, 512, 512, 512, nullptr);
  knc_attn_k<<<BN/256, 256, 0, stream>>>(pv, cxyz, attn, idx2);
  local_gather_k<<<(unsigned)cdiv_((long)BN*512,256), 256, 0, stream>>>(pvgW, attn, idx2, lcl);
  GEMM(0,0,0, lcl, wt+WT_FLP, pr[P_FLP_B], la, BN, 1024, 512, 512, nullptr);
  layernorm_act_k<<<BN, 256, 0, stream>>>(la, pr[P_FLP_G], pr[P_FLP_BE], 1024, 1);
  gctx_k<<<(unsigned)cdiv_((long)BN*1024,256), 256, 0, stream>>>(gmx, f2, gctx);

  // gate: relu(gctx@W1a + la@W1b + b1) -> sigmoid-gate fused into fg_w2 GEMM
  GEMM(0,0,0, gctx, wt+WT_FG1,        nullptr,      tg, BN, 64, 1024, 2048, nullptr);
  GEMM(1,1,0, la,   wt+WT_FG1+1024,   pr[P_FG_B1],  tg, BN, 64, 1024, 2048, nullptr);
  GEMM(0,0,1, tg,   wt+WT_FG2,        pr[P_FG_B2],  gctx, BN, 1024, 64, 64, la);
  // gctx now holds fused = gctx + sigmoid(.)*la

  GEMM(0,0,0, gctx, wt+WT_L4, pr[P_L4_B], la, BN, 1024, 1024, 1024, nullptr);
  layernorm_act_k<<<BN, 256, 0, stream>>>(la, pr[P_L4_G], pr[P_L4_BE], 1024, 1);
  GEMM(2,0,0, la,   wt+WT_L5, pr[P_L5_B], gctx, BN, 512, 1024, 1024, nullptr);
  GEMM(2,0,0, gctx, wt+WT_L6, pr[P_L6_B], la,   BN, 256,  512,  512, nullptr);
  GEMM(2,0,0, la,   wt+WT_L7, pr[P_L7_B], gctx, BN,  64,  256,  256, nullptr);
  head_k<<<(unsigned)cdiv_((long)BN*3,256), 256, 0, stream>>>(gctx, pr[P_LG_W], pr[P_LG_B], outp);

  #undef GEMM
}
